// SSVLI_SigLipLoss_2078764171850
// MI455X (gfx1250) — matrix operand order repaired and checked by reading, not compiled, not run on hardware
//
#include <hip/hip_runtime.h>

// ---------------------------------------------------------------------------
// SigLIP loss for MI455X (gfx1250): bf16 WMMA GEMM (hi/lo f32-recovery) with
// TDM (tensor_load_to_lds) double-buffered LDS staging.
// ---------------------------------------------------------------------------

typedef __attribute__((ext_vector_type(16))) __bf16        v16bf;
typedef __attribute__((ext_vector_type(8)))  float         v8f;
typedef __attribute__((ext_vector_type(4)))  unsigned int  u32x4;
typedef __attribute__((ext_vector_type(8)))  unsigned int  u32x8;

#define N_ROWS 8192
#define DIM    768
#define TILE   128   // logits tile per workgroup (128x128)
#define TK     32    // bf16 WMMA K-chunk
#define LSTR   48    // LDS row stride in elements (96B = 64B row + 32B TDM pad)

union FragU {
  v16bf v;
  uint4 q[2];
};

__device__ __forceinline__ uint4 lds_ld_b128(const __bf16* p) {
  return *reinterpret_cast<const uint4*>(p);
}

// A fragment (16x32 bf16): lanes 0-15: K 0..7 & 16..23 ; lanes 16-31: K 8..15 & 24..31
__device__ __forceinline__ v16bf frag_a(const __bf16* row, int lane) {
  FragU f;
  const int off = (lane & 16) >> 1;       // 0 | 8
  f.q[0] = lds_ld_b128(row + off);
  f.q[1] = lds_ld_b128(row + off + 16);
  return f.v;
}

// B fragment (32x16 bf16, one column per lane == one row of t):
// lanes 0-15: K 0..7 & 16..23 ; lanes 16-31: K 8..15 & 24..31
__device__ __forceinline__ v16bf frag_b(const __bf16* row, int lane) {
  FragU f;
  const int off = (lane & 16) >> 1;       // 0 | 8
  f.q[0] = lds_ld_b128(row + off);
  f.q[1] = lds_ld_b128(row + off + 16);
  return f.v;
}

// TDM: DMA one 128-row x 32-elem bf16 tile (row stride DIM) from global memory
// into LDS at lds dest, inserting 32B of padding after every 64B row so rows
// land at a 96B (LSTR) stride. 2D tensor -> descriptor groups 0 and 1 only.
__device__ __forceinline__ void tdm_load_2d(__bf16* lds, const __bf16* g) {
  const unsigned lds_off = (unsigned)(uintptr_t)lds;        // addr[31:0] == LDS offset
  const unsigned long long ga = (unsigned long long)(uintptr_t)g;
  u32x4 g0;
  g0.x = 1u;                                                 // count=1, user mode
  g0.y = lds_off;                                            // lds_addr
  g0.z = (unsigned)ga;                                       // global_addr[31:0]
  g0.w = ((unsigned)(ga >> 32) & 0x01FFFFFFu) | (2u << 30);  // global_addr[56:32] | type=2
  u32x8 g1;
  g1[0] = (1u << 16)        // data_size = 2 bytes
        | (1u << 20)        // pad_enable
        | (3u << 22)        // pad_interval: 16 DWORDs (64B) between pads
        | (7u << 25);       // pad_amount: 8 DWORDs (32B)
  g1[1] = ((unsigned)DIM & 0xFFFFu) << 16;                   // tensor_dim0[15:0]
  g1[2] = ((unsigned)DIM >> 16) | (((unsigned)N_ROWS & 0xFFFFu) << 16); // dim0 hi | dim1 lo
  g1[3] = ((unsigned)N_ROWS >> 16) | ((unsigned)TK << 16);   // dim1 hi | tile_dim0=32
  g1[4] = (unsigned)TILE;                                    // tile_dim1=128 | tile_dim2=0
  g1[5] = (unsigned)DIM;                                     // tensor_dim0_stride[31:0]
  g1[6] = 0u;                                                // stride hi | dim1_stride lo
  g1[7] = 0u;
  asm volatile("tensor_load_to_lds %0, %1" :: "s"(g0), "s"(g1) : "memory");
}

// -log_sigmoid(-z) = softplus(z) = log(1+e^z), numerically stable
__device__ __forceinline__ float softplusf(float z) {
  return fmaxf(z, 0.0f) + __logf(1.0f + __expf(-fabsf(z)));
}

// Monotone float->u32 map so larger float == larger unsigned; tie -> smaller col
__device__ __forceinline__ unsigned long long pack_key(float v, unsigned col) {
  unsigned u = __float_as_uint(v);
  unsigned m = (u & 0x80000000u) ? ~u : (u | 0x80000000u);
  return ((unsigned long long)m << 32) | (unsigned)(~col);
}

// ---------------------------------------------------------------------------
__global__ void SSVLI_init_kernel(unsigned long long* __restrict__ keys,
                                  double* __restrict__ loss) {
  const int i = blockIdx.x * 256 + threadIdx.x;
  if (i < N_ROWS) keys[i] = 0ull;
  if (i == 0) *loss = 0.0;
}

// ---------------------------------------------------------------------------
// Normalize one row (768 = 256*3 elems) and split into bf16 hi/lo streams.
__global__ __launch_bounds__(256)
void SSVLI_norm_split_kernel(const float* __restrict__ vin,
                             const float* __restrict__ tin,
                             __bf16* __restrict__ vhi, __bf16* __restrict__ vlo,
                             __bf16* __restrict__ thi, __bf16* __restrict__ tlo) {
  __shared__ float red[256];
  const int row = blockIdx.x;
  const float* src = blockIdx.y ? tin : vin;
  __bf16* hi = blockIdx.y ? thi : vhi;
  __bf16* lo = blockIdx.y ? tlo : vlo;

  const size_t base = (size_t)row * DIM;
  const int t = threadIdx.x;
  float x0 = src[base + t];
  float x1 = src[base + t + 256];
  float x2 = src[base + t + 512];

  red[t] = x0 * x0 + x1 * x1 + x2 * x2;
  __syncthreads();
  for (int off = 128; off > 0; off >>= 1) {
    if (t < off) red[t] += red[t + off];
    __syncthreads();
  }
  const float inv = rsqrtf(red[0]);

  float n0 = x0 * inv, n1 = x1 * inv, n2 = x2 * inv;
  __bf16 h0 = (__bf16)n0, h1 = (__bf16)n1, h2 = (__bf16)n2;
  hi[base + t]       = h0;  lo[base + t]       = (__bf16)(n0 - (float)h0);
  hi[base + t + 256] = h1;  lo[base + t + 256] = (__bf16)(n1 - (float)h1);
  hi[base + t + 512] = h2;  lo[base + t + 512] = (__bf16)(n2 - (float)h2);
}

// ---------------------------------------------------------------------------
// 128x128 logits tile: 8 waves, each wave owns 32 rows x 64 cols (2x4 WMMA
// subtiles). TDM double-buffers the K-chunks into LDS; 3 bf16 WMMAs per
// subtile per K-chunk recover ~f32 accuracy from the hi/lo split.
__global__ __launch_bounds__(256)
void SSVLI_gemm_loss_kernel(const __bf16* __restrict__ vhi, const __bf16* __restrict__ vlo,
                            const __bf16* __restrict__ thi, const __bf16* __restrict__ tlo,
                            const float* __restrict__ lls,
                            unsigned long long* __restrict__ rowKeys,
                            double* __restrict__ lossAcc) {
  // [stage][stream][tile]: streams 0=Ahi 1=Alo 2=Bhi 3=Blo. 96KB of the WGP's 320KB.
  __shared__ __bf16 lbuf[2][4][TILE * LSTR];

  const int tid   = threadIdx.x;
  const int lane  = tid & 31;
  const int wave  = tid >> 5;
  const int waveM = wave >> 1;            // 0..3 -> 32 rows each
  const int waveN = wave & 1;             // 0..1 -> 64 cols each
  const int tileM = blockIdx.y * TILE;
  const int tileN = blockIdx.x * TILE;
  const int mr    = lane & 15;

  v8f acc[2][4];
  v8f zero = {};
#pragma unroll
  for (int i = 0; i < 2; ++i)
#pragma unroll
    for (int j = 0; j < 4; ++j) acc[i][j] = zero;

  // Prologue: wave 0 kicks off the DMA for K-chunk 0 into stage 0.
  if (wave == 0) {
    tdm_load_2d(&lbuf[0][0][0], vhi + (size_t)tileM * DIM);
    tdm_load_2d(&lbuf[0][1][0], vlo + (size_t)tileM * DIM);
    tdm_load_2d(&lbuf[0][2][0], thi + (size_t)tileN * DIM);
    tdm_load_2d(&lbuf[0][3][0], tlo + (size_t)tileN * DIM);
  }

  int p = 0;
  for (int kc = 0; kc < DIM; kc += TK) {
    __builtin_amdgcn_s_wait_tensorcnt(0);   // wave 0: stage p landed (no-op elsewhere)
    __syncthreads();                        // publish stage p; all reads of p^1 drained
    if (wave == 0 && kc + TK < DIM) {       // overlap: DMA next chunk into stage p^1
      const int kn = kc + TK;
      tdm_load_2d(&lbuf[p ^ 1][0][0], vhi + (size_t)tileM * DIM + kn);
      tdm_load_2d(&lbuf[p ^ 1][1][0], vlo + (size_t)tileM * DIM + kn);
      tdm_load_2d(&lbuf[p ^ 1][2][0], thi + (size_t)tileN * DIM + kn);
      tdm_load_2d(&lbuf[p ^ 1][3][0], tlo + (size_t)tileN * DIM + kn);
    }

    v16bf aH[2], aL[2], bH[4], bL[4];
#pragma unroll
    for (int i = 0; i < 2; ++i) {
      const int r = (waveM * 32 + i * 16 + mr) * LSTR;
      aH[i] = frag_a(&lbuf[p][0][r], lane);
      aL[i] = frag_a(&lbuf[p][1][r], lane);
    }
#pragma unroll
    for (int j = 0; j < 4; ++j) {
      const int c = (waveN * 64 + j * 16 + mr) * LSTR;
      bH[j] = frag_b(&lbuf[p][2][c], lane);
      bL[j] = frag_b(&lbuf[p][3][c], lane);
    }
#pragma unroll
    for (int i = 0; i < 2; ++i)
#pragma unroll
      for (int j = 0; j < 4; ++j) {
        acc[i][j] = __builtin_amdgcn_wmma_f32_16x16x32_bf16(
            false, aH[i], false, bH[j], (short)0, acc[i][j], false, false);
        acc[i][j] = __builtin_amdgcn_wmma_f32_16x16x32_bf16(
            false, aH[i], false, bL[j], (short)0, acc[i][j], false, false);
        acc[i][j] = __builtin_amdgcn_wmma_f32_16x16x32_bf16(
            false, aL[i], false, bH[j], (short)0, acc[i][j], false, false);
      }
    p ^= 1;
  }

  // ---- epilogue: softplus loss + per-row argmax ----
  // C/D layout: lane 0-15 -> col=lane, row=vgpr ; lane 16-31 -> col=lane-16, row=vgpr+8
  const float scale  = __expf(lls[0]);
  const int halfUp   = (lane >> 4) << 3;
  const int colLane  = lane & 15;
  float lsum = 0.0f;

#pragma unroll
  for (int i = 0; i < 2; ++i) {
#pragma unroll
    for (int r = 0; r < 8; ++r) {
      const int grow = tileM + waveM * 32 + i * 16 + r + halfUp;
      float best = -3.0e38f;
      int bestc = 0;
#pragma unroll
      for (int j = 0; j < 4; ++j) {
        const int gcol = tileN + waveN * 64 + j * 16 + colLane;
        const float x = scale * acc[i][j][r];
        lsum += softplusf(grow == gcol ? -x : x);   // -log_sigmoid(label*x)
        if (x > best) { best = x; bestc = gcol; }
      }
      // reduce across the 16 lanes holding this row (xor masks stay in-group)
#pragma unroll
      for (int off = 8; off > 0; off >>= 1) {
        const float ov = __shfl_xor(best, off, 32);
        const int   oc = __shfl_xor(bestc, off, 32);
        if (ov > best || (ov == best && oc < bestc)) { best = ov; bestc = oc; }
      }
      if (colLane == 0)
        atomicMax(&rowKeys[grow], pack_key(best, (unsigned)bestc));
    }
  }

  // In-wave loss reduction, one f64 atomic per wave.
#pragma unroll
  for (int off = 16; off > 0; off >>= 1)
    lsum += __shfl_xor(lsum, off, 32);
  if (lane == 0)
    __hip_atomic_fetch_add(lossAcc, (double)lsum,
                           __ATOMIC_RELAXED, __HIP_MEMORY_SCOPE_AGENT);
}

// ---------------------------------------------------------------------------
__global__ __launch_bounds__(256)
void SSVLI_finalize_kernel(const unsigned long long* __restrict__ keys,
                           const double* __restrict__ loss,
                           float* __restrict__ out) {
  __shared__ int red[256];
  int cnt = 0;
  for (int r = threadIdx.x; r < N_ROWS; r += 256) {
    const unsigned col = ~(unsigned)(keys[r] & 0xFFFFFFFFull);
    if (col == (unsigned)r) ++cnt;
  }
  red[threadIdx.x] = cnt;
  __syncthreads();
  for (int off = 128; off > 0; off >>= 1) {
    if (threadIdx.x < off) red[threadIdx.x] += red[threadIdx.x + off];
    __syncthreads();
  }
  if (threadIdx.x == 0) {
    out[0] = (float)(*loss / (double)N_ROWS);
    out[1] = 100.0f * (float)red[0] / (float)N_ROWS;
  }
}

// ---------------------------------------------------------------------------
extern "C" void kernel_launch(void* const* d_in, const int* in_sizes, int n_in,
                              void* d_out, int out_size, void* d_ws, size_t ws_size,
                              hipStream_t stream) {
  (void)in_sizes; (void)n_in; (void)out_size; (void)ws_size;
  const float* vin = (const float*)d_in[0];
  const float* tin = (const float*)d_in[1];
  const float* lls = (const float*)d_in[2];
  float* out = (float*)d_out;

  char* ws = (char*)d_ws;
  const size_t ND = (size_t)N_ROWS * DIM;
  __bf16* vhi = (__bf16*)(ws);
  __bf16* vlo = (__bf16*)(ws + 2 * ND);
  __bf16* thi = (__bf16*)(ws + 4 * ND);
  __bf16* tlo = (__bf16*)(ws + 6 * ND);
  unsigned long long* keys = (unsigned long long*)(ws + 8 * ND);
  double* loss = (double*)(ws + 8 * ND + (size_t)N_ROWS * sizeof(unsigned long long));

  SSVLI_init_kernel<<<(N_ROWS + 255) / 256, 256, 0, stream>>>(keys, loss);
  SSVLI_norm_split_kernel<<<dim3(N_ROWS, 2), 256, 0, stream>>>(vin, tin, vhi, vlo, thi, tlo);
  SSVLI_gemm_loss_kernel<<<dim3(N_ROWS / TILE, N_ROWS / TILE), 256, 0, stream>>>(
      vhi, vlo, thi, tlo, lls, keys, loss);
  SSVLI_finalize_kernel<<<1, 256, 0, stream>>>(keys, loss, out);
}
